// WaveNetPredictor_80101140070453
// MI455X (gfx1250) — matrix operand order repaired and checked by reading, compiled, not run on hardware
//
#include <hip/hip_runtime.h>

// ---------------------------------------------------------------------------
// WaveNet forward, MI455X (gfx1250, wave32, WMMA f32_16x16x32_f16).
// Only the last 1024 timesteps (receptive field) are computed.
// Workspace: hA (8MB) | hB (8MB) | skip (256KB)   -> ~16.5 MB of d_ws.
// All global loads are unconditional & in-bounds; boundary masking is done
// with VALU selects (no exec-mask branches).
// ---------------------------------------------------------------------------

typedef __attribute__((ext_vector_type(16))) _Float16 v16h;
typedef __attribute__((ext_vector_type(8)))  _Float16 v8h;
typedef __attribute__((ext_vector_type(8)))  float    v8f;

#define WIN   1024   // receptive-field window
#define CCH   64     // residual channels
#define SCH   128    // skip channels
#define TFULL 4096
#define NB    32
#define DIN   50
#define NLAY  10

// A-operand K order (contiguous, matches B operands):
// lane group g = lane>>4; half h -> K = h + 16g
// => per lane: one contiguous run of 16: [kt*32+16g, +16)
__device__ __forceinline__ int klocA(int h, int g) {
    return h + 16 * g;
}

__device__ __forceinline__ v8f wmma_f16(v16h a, v16h b, v8f c) {
    return __builtin_amdgcn_wmma_f32_16x16x32_f16(
        /*neg_a=*/false, a, /*neg_b=*/false, b,
        /*c_mod=*/(short)0, c, /*reuse_a=*/false, /*reuse_b=*/false);
}

// 16 consecutive f32 (16B aligned) -> B-operand fragment
__device__ __forceinline__ v16h load_bfrag_f32(const float* __restrict__ p) {
    float4 a = *(const float4*)(p + 0);
    float4 b = *(const float4*)(p + 4);
    float4 c = *(const float4*)(p + 8);
    float4 d = *(const float4*)(p + 12);
    v16h r;
    r[0]  = (_Float16)a.x; r[1]  = (_Float16)a.y; r[2]  = (_Float16)a.z; r[3]  = (_Float16)a.w;
    r[4]  = (_Float16)b.x; r[5]  = (_Float16)b.y; r[6]  = (_Float16)b.z; r[7]  = (_Float16)b.w;
    r[8]  = (_Float16)c.x; r[9]  = (_Float16)c.y; r[10] = (_Float16)c.z; r[11] = (_Float16)c.w;
    r[12] = (_Float16)d.x; r[13] = (_Float16)d.y; r[14] = (_Float16)d.z; r[15] = (_Float16)d.w;
    return r;
}

// same, but zero the whole fragment when !valid (selects, no branches)
__device__ __forceinline__ v16h load_bfrag_f32_masked(const float* __restrict__ p, bool valid) {
    float4 a = *(const float4*)(p + 0);
    float4 b = *(const float4*)(p + 4);
    float4 c = *(const float4*)(p + 8);
    float4 d = *(const float4*)(p + 12);
    float fv[16] = {a.x,a.y,a.z,a.w, b.x,b.y,b.z,b.w, c.x,c.y,c.z,c.w, d.x,d.y,d.z,d.w};
    v16h r;
    #pragma unroll
    for (int i = 0; i < 16; ++i) r[i] = (_Float16)(valid ? fv[i] : 0.f);
    return r;
}

// A-fragment gather from a dense f32 row (all 32 K-columns valid).
// rowbase must be 32B aligned at the run offsets (row stride multiple of 8).
__device__ __forceinline__ v16h load_afrag_rows(const float* __restrict__ rowbase, int kt, int g) {
    const float* p1 = rowbase + kt * 32 + 16 * g;
    const float* p2 = rowbase + kt * 32 + 16 * g + 8;
    float4 a = *(const float4*)(p1 + 0);
    float4 b = *(const float4*)(p1 + 4);
    float4 c = *(const float4*)(p2 + 0);
    float4 d = *(const float4*)(p2 + 4);
    v16h r;
    r[0]  = (_Float16)a.x; r[1]  = (_Float16)a.y; r[2]  = (_Float16)a.z; r[3]  = (_Float16)a.w;
    r[4]  = (_Float16)b.x; r[5]  = (_Float16)b.y; r[6]  = (_Float16)b.z; r[7]  = (_Float16)b.w;
    r[8]  = (_Float16)c.x; r[9]  = (_Float16)c.y; r[10] = (_Float16)c.z; r[11] = (_Float16)c.w;
    r[12] = (_Float16)d.x; r[13] = (_Float16)d.y; r[14] = (_Float16)d.z; r[15] = (_Float16)d.w;
    return r;
}

__device__ __forceinline__ void load_bias8(const float* __restrict__ p, v8f& acc) {
    float4 a = *(const float4*)(p + 0);
    float4 b = *(const float4*)(p + 4);
    acc[0] = a.x; acc[1] = a.y; acc[2] = a.z; acc[3] = a.w;
    acc[4] = b.x; acc[5] = b.y; acc[6] = b.z; acc[7] = b.w;
}

// ---------------------------------------------------------------------------
// Kernel 1: h0[b][t][c] = W_in @ x + b_in   (last WIN timesteps), zero skipbuf
// ---------------------------------------------------------------------------
__global__ void __launch_bounds__(128) wn_embed(
    const float* __restrict__ x, const float* __restrict__ Win,
    const float* __restrict__ bin, float* __restrict__ hA,
    float* __restrict__ skipbuf)
{
    // zero the skip accumulator (exactly 512*128 = 65536 = NB*SCH*16 threads)
    int zi = blockIdx.x * blockDim.x + threadIdx.x;
    if (zi < NB * SCH * 16) skipbuf[zi] = 0.f;

    const int l = threadIdx.x & 31, wave = threadIdx.x >> 5;
    const int b = blockIdx.x >> 4, tgrp = blockIdx.x & 15;
    const int n = l & 15, g = l >> 4;
    const int t = tgrp * 64 + wave * 16 + n;
    const long Tg = (long)(TFULL - WIN) + t;
    const float* xrow = x + ((long)b * TFULL + Tg) * DIN;  // 8B-aligned only

    // B fragments from x (K = input features, zero-padded 50 -> 64)
    v16h xb[2];
    {   // kt=0: K = 16g + h, 0..31 all valid
        float2 q[8];
        #pragma unroll
        for (int j = 0; j < 8; ++j) q[j] = *(const float2*)(xrow + 16 * g + 2 * j);
        #pragma unroll
        for (int h = 0; h < 16; ++h)
            xb[0][h] = (_Float16)((h & 1) ? q[h >> 1].y : q[h >> 1].x);
    }
    {   // kt=1: K = 32 + 16g + h; g==0 -> 32..47 valid; g==1 -> only 48,49
        float2 q[8];
        #pragma unroll
        for (int j = 0; j < 8; ++j) q[j] = *(const float2*)(xrow + 32 + 2 * j);
        float2 p48 = *(const float2*)(xrow + 48);   // always in-bounds
        #pragma unroll
        for (int h = 0; h < 16; ++h) {
            float full = (h & 1) ? q[h >> 1].y : q[h >> 1].x;
            float part = (h == 0) ? p48.x : (h == 1) ? p48.y : 0.f;
            xb[1][h] = (_Float16)((g == 0) ? full : part);
        }
    }

    #pragma unroll
    for (int mt = 0; mt < 4; ++mt) {
        v8f acc;
        load_bias8(bin + mt * 16 + 8 * g, acc);
        const int m = mt * 16 + (l & 15);
        const float* wrow = Win + m * DIN;          // 8B-aligned (50*4=200)
        {   // kt=0: run at 16g (0..31), all valid
            v16h a;
            float2 q1[4], q2[4];
            #pragma unroll
            for (int j = 0; j < 4; ++j) {
                q1[j] = *(const float2*)(wrow + 16 * g + 2 * j);
                q2[j] = *(const float2*)(wrow + 16 * g + 8 + 2 * j);
            }
            #pragma unroll
            for (int h = 0; h < 8; ++h) {
                a[h]     = (_Float16)((h & 1) ? q1[h >> 1].y : q1[h >> 1].x);
                a[h + 8] = (_Float16)((h & 1) ? q2[h >> 1].y : q2[h >> 1].x);
            }
            acc = wmma_f16(a, xb[0], acc);
        }
        {   // kt=1: K = 32+16g+h; g==0 -> 32..47 valid; g==1 -> only 48,49
            v16h a;
            float2 q1[4], q2[4];
            #pragma unroll
            for (int j = 0; j < 4; ++j) {
                q1[j] = *(const float2*)(wrow + 32 + 2 * j);
                q2[j] = *(const float2*)(wrow + 40 + 2 * j);
            }
            float2 p48 = *(const float2*)(wrow + 48);  // always in-bounds
            #pragma unroll
            for (int h = 0; h < 8; ++h) {
                float f1 = (h & 1) ? q1[h >> 1].y : q1[h >> 1].x;
                float f2 = (h & 1) ? q2[h >> 1].y : q2[h >> 1].x;
                float part = (h == 0) ? p48.x : (h == 1) ? p48.y : 0.f;
                a[h] = (_Float16)((g == 0) ? f1 : part);
                a[h + 8] = (_Float16)((g == 0) ? f2 : 0.f);
            }
            acc = wmma_f16(a, xb[1], acc);
        }
        float* dst = hA + (long)(b * WIN + t) * CCH + mt * 16 + 8 * g;
        *(float4*)(dst + 0) = make_float4(acc[0], acc[1], acc[2], acc[3]);
        *(float4*)(dst + 4) = make_float4(acc[4], acc[5], acc[6], acc[7]);
    }
}

// ---------------------------------------------------------------------------
// Kernel 2 (x10): one dilated WaveNet layer.
//   y = Wd[:,:,0]@h[t-d] + Wd[:,:,1]@h[t] + bd
//   z = tanh(y[64:]) * sigmoid(y[:64])        (z=0 for t<d)
//   hout = hin + Wr@z + br
//   skip[b,:,lastTile] += Ws@z + bs           (only last 16 columns)
// ---------------------------------------------------------------------------
__global__ void __launch_bounds__(128) wn_layer(
    const float* __restrict__ hin, float* __restrict__ hout,
    float* __restrict__ skipbuf,
    const float* __restrict__ Wd, const float* __restrict__ bd,
    const float* __restrict__ Wr, const float* __restrict__ br,
    const float* __restrict__ Ws, const float* __restrict__ bs,
    int d)
{
    __shared__ _Float16 wlds[40 * 512];      // 32 Wd frags + 8 Wr frags (40 KB)
    __shared__ _Float16 zlds[4 * 16 * CCH];  // per-wave z tile [n][c] (8 KB)

    // Cooperative stage of weights, pre-swizzled into A-fragment layout.
    for (int e = threadIdx.x; e < 40 * 512; e += 128) {
        int fr = e >> 9, rr = e & 511, lw = rr >> 4, h = rr & 15;
        int gg = lw >> 4, mrow = lw & 15;
        float v;
        if (fr < 32) {                       // Wd: frag = tap*16 + mt*2 + kt
            int tap = fr >> 4, mk = fr & 15, mt = mk >> 1, kt = mk & 1;
            int m = mt * 16 + mrow;
            int c = kt * 32 + klocA(h, gg);
            v = Wd[(m * CCH + c) * 2 + tap];
        } else {                             // Wr: frag = mt*2 + kt
            int f2 = fr - 32, mt = f2 >> 1, kt = f2 & 1;
            int m = mt * 16 + mrow;
            int c = kt * 32 + klocA(h, gg);
            v = Wr[m * CCH + c];
        }
        wlds[e] = (_Float16)v;
    }
    __syncthreads();

    const int l = threadIdx.x & 31, wave = threadIdx.x >> 5;
    const int b = blockIdx.x >> 4, tgrp = blockIdx.x & 15;
    const int n = l & 15, g = l >> 4;
    const int t = tgrp * 64 + wave * 16 + n;
    const int tl = t - d;
    const int tlc = (tl >= 0) ? tl : 0;      // clamped (address always valid)
    const bool lv = (tl >= 0);

    // B fragments: current and lagged h columns (lag masked, not branched)
    v16h cur[2], lag[2];
    #pragma unroll
    for (int kt = 0; kt < 2; ++kt) {
        cur[kt] = load_bfrag_f32(hin + (long)(b * WIN + t) * CCH + kt * 32 + 16 * g);
        lag[kt] = load_bfrag_f32_masked(
            hin + (long)(b * WIN + tlc) * CCH + kt * 32 + 16 * g, lv);
    }

    // y: 8 M-tiles (rows 0..63 gate, 64..127 filt)
    v8f y[8];
    #pragma unroll
    for (int mt = 0; mt < 8; ++mt) {
        v8f acc;
        load_bias8(bd + mt * 16 + 8 * g, acc);
        #pragma unroll
        for (int kt = 0; kt < 2; ++kt) {
            v16h a0 = *(const v16h*)&wlds[(0 * 16 + mt * 2 + kt) * 512 + l * 16];
            acc = wmma_f16(a0, lag[kt], acc);
            v16h a1 = *(const v16h*)&wlds[(1 * 16 + mt * 2 + kt) * 512 + l * 16];
            acc = wmma_f16(a1, cur[kt], acc);
        }
        y[mt] = acc;
    }

    // z = tanh(filt)*sigmoid(gate); zero columns with t<d. D-layout -> LDS [n][c]
    #pragma unroll
    for (int mt = 0; mt < 4; ++mt) {
        v8h zh;
        #pragma unroll
        for (int r = 0; r < 8; ++r) {
            float gate = y[mt][r], filt = y[mt + 4][r];
            float zv = tanhf(filt) * (1.f / (1.f + __expf(-gate)));
            zh[r] = (_Float16)((t >= d) ? zv : 0.f);
        }
        *(v8h*)&zlds[wave * (16 * CCH) + n * CCH + mt * 16 + 8 * g] = zh;
    }
    __syncthreads();

    // reload z as B-operand fragments (contiguous K per lane)
    v16h zb[2];
    #pragma unroll
    for (int kt = 0; kt < 2; ++kt)
        zb[kt] = *(const v16h*)&zlds[wave * (16 * CCH) + n * CCH + kt * 32 + 16 * g];

    // residual: hout = hin + Wr @ z + br
    #pragma unroll
    for (int mt = 0; mt < 4; ++mt) {
        v8f acc;
        const float* ph = hin + (long)(b * WIN + t) * CCH + mt * 16 + 8 * g;
        float4 h0 = *(const float4*)(ph + 0);
        float4 h1 = *(const float4*)(ph + 4);
        float4 b0 = *(const float4*)(br + mt * 16 + 8 * g);
        float4 b1v = *(const float4*)(br + mt * 16 + 8 * g + 4);
        acc[0] = h0.x + b0.x;  acc[1] = h0.y + b0.y;
        acc[2] = h0.z + b0.z;  acc[3] = h0.w + b0.w;
        acc[4] = h1.x + b1v.x; acc[5] = h1.y + b1v.y;
        acc[6] = h1.z + b1v.z; acc[7] = h1.w + b1v.w;
        #pragma unroll
        for (int kt = 0; kt < 2; ++kt) {
            v16h a = *(const v16h*)&wlds[(32 + mt * 2 + kt) * 512 + l * 16];
            acc = wmma_f16(a, zb[kt], acc);
        }
        float* po = hout + (long)(b * WIN + t) * CCH + mt * 16 + 8 * g;
        *(float4*)(po + 0) = make_float4(acc[0], acc[1], acc[2], acc[3]);
        *(float4*)(po + 4) = make_float4(acc[4], acc[5], acc[6], acc[7]);
    }

    // skip accumulation: only the wave owning t in [WIN-16, WIN) for this batch
    if (tgrp == 15 && wave == 3) {
        #pragma unroll
        for (int mt = 0; mt < 8; ++mt) {
            v8f acc;
            load_bias8(bs + mt * 16 + 8 * g, acc);
            #pragma unroll
            for (int r = 0; r < 8; ++r)
                acc[r] += skipbuf[(b * SCH + mt * 16 + 8 * g + r) * 16 + n];
            #pragma unroll
            for (int kt = 0; kt < 2; ++kt) {
                v16h a = load_afrag_rows(Ws + (mt * 16 + (l & 15)) * CCH, kt, g);
                acc = wmma_f16(a, zb[kt], acc);
            }
            #pragma unroll
            for (int r = 0; r < 8; ++r)
                skipbuf[(b * SCH + mt * 16 + 8 * g + r) * 16 + n] = acc[r];
        }
    }
}

// ---------------------------------------------------------------------------
// Kernel 3: head. o0 = relu(skip[:,:,last]); o1 = relu(W1@o0 + b1);
//           out[b] = W2@o1 + b2.   Single block, 128 threads.
// ---------------------------------------------------------------------------
__global__ void __launch_bounds__(128) wn_head(
    const float* __restrict__ skipbuf,
    const float* __restrict__ W1, const float* __restrict__ b1,
    const float* __restrict__ W2, const float* __restrict__ b2,
    float* __restrict__ out)
{
    __shared__ _Float16 o0[SCH * NB];  // [s][b]
    __shared__ float    o1[SCH * NB];  // [u][b]

    for (int e = threadIdx.x; e < SCH * NB; e += 128) {
        int s = e >> 5, b = e & 31;
        float v = skipbuf[(b * SCH + s) * 16 + 15];   // column t = WIN-1
        o0[s * NB + b] = (_Float16)fmaxf(v, 0.f);
    }
    __syncthreads();

    const int l = threadIdx.x & 31, wave = threadIdx.x >> 5;
    const int n = l & 15, g = l >> 4;
    if (wave < 2) {   // two N-tiles of 16 batches; GEMM M=128,K=128,N=32
        int bcol = wave * 16 + n;
        #pragma unroll
        for (int mt = 0; mt < 8; ++mt) {
            v8f acc;
            load_bias8(b1 + mt * 16 + 8 * g, acc);
            #pragma unroll
            for (int kt = 0; kt < 4; ++kt) {
                v16h a = load_afrag_rows(W1 + (mt * 16 + (l & 15)) * SCH, kt, g);
                v16h bb;
                #pragma unroll
                for (int h = 0; h < 16; ++h)
                    bb[h] = o0[(kt * 32 + 16 * g + h) * NB + bcol];
                acc = wmma_f16(a, bb, acc);
            }
            #pragma unroll
            for (int r = 0; r < 8; ++r)
                o1[(mt * 16 + 8 * g + r) * NB + bcol] = fmaxf(acc[r], 0.f);
        }
    }
    __syncthreads();

    if (threadIdx.x < NB) {
        int b = threadIdx.x;
        float acc = b2[0];
        for (int m = 0; m < SCH; ++m) acc += W2[m] * o1[m * NB + b];
        out[b] = acc;
    }
}

// ---------------------------------------------------------------------------
extern "C" void kernel_launch(void* const* d_in, const int* in_sizes, int n_in,
                              void* d_out, int out_size, void* d_ws, size_t ws_size,
                              hipStream_t stream)
{
    (void)in_sizes; (void)n_in; (void)out_size; (void)ws_size;
    const float* x   = (const float*)d_in[0];
    const float* Win = (const float*)d_in[1];
    const float* bin = (const float*)d_in[2];
    const float* Wd  = (const float*)d_in[3];
    const float* bd  = (const float*)d_in[4];
    const float* Wr  = (const float*)d_in[5];
    const float* br  = (const float*)d_in[6];
    const float* Ws  = (const float*)d_in[7];
    const float* bs  = (const float*)d_in[8];
    const float* W1  = (const float*)d_in[9];
    const float* b1  = (const float*)d_in[10];
    const float* W2  = (const float*)d_in[11];
    const float* b2  = (const float*)d_in[12];
    float* out = (float*)d_out;

    char* ws = (char*)d_ws;
    const size_t hbytes = (size_t)NB * WIN * CCH * sizeof(float);  // 8 MB
    float* hA   = (float*)(ws);
    float* hB   = (float*)(ws + hbytes);
    float* skip = (float*)(ws + 2 * hbytes);                       // 256 KB

    wn_embed<<<512, 128, 0, stream>>>(x, Win, bin, hA, skip);

    float* hin = hA;
    float* hout = hB;
    for (int i = 0; i < NLAY; ++i) {
        wn_layer<<<512, 128, 0, stream>>>(
            hin, hout, skip,
            Wd + (size_t)i * 2 * CCH * CCH * 2,   // [2C][C][2] per layer
            bd + (size_t)i * 2 * CCH,
            Wr + (size_t)i * CCH * CCH,
            br + (size_t)i * CCH,
            Ws + (size_t)i * SCH * CCH,
            bs + (size_t)i * SCH,
            1 << i);
        float* tmp = hin; hin = hout; hout = tmp;
    }

    wn_head<<<1, 128, 0, stream>>>(skip, W1, b1, W2, b2, out);
}
